// MODEL_83399674954070
// MI455X (gfx1250) — matrix operand order repaired and checked by reading, not compiled, not run on hardware
//
#include <hip/hip_runtime.h>

#define STACKT 10
#define TIER 4
#define NLAYER 4

typedef __attribute__((ext_vector_type(16))) __bf16 bf16x16;
typedef __attribute__((ext_vector_type(8)))  float  f32x8;
typedef __attribute__((ext_vector_type(8)))  unsigned short u16x8;
typedef __attribute__((ext_vector_type(16))) unsigned short u16x16;

// ---------------- workspace layout (bytes) ----------------
#define WIH_HALVES  262144   // [L=4][dir=2][kt=4][nt=16][lane=32][16]
#define WHH_HALVES  131072   // [4][2][kt=2][nt=16][32][16]
#define FC1_HALVES  8192     // [kt=4][nt=4][32][16]
#define FC2_HALVES  3072     // [kt=2][nt=3][32][16]
#define FC3_HALVES  2048     // [kt=2][nt=2][32][16]  (K padded 48->64 with zeros)
#define BIASN       2048     // [4][2][256] f32  (b_ih + b_hh)

#define WIH_OFF  0
#define WHH_OFF  (WIH_OFF + WIH_HALVES*2)
#define BIAS_OFF (WHH_OFF + WHH_HALVES*2)
#define FC1_OFF  (BIAS_OFF + BIASN*4)
#define FC2_OFF  (FC1_OFF + FC1_HALVES*2)
#define FC3_OFF  (FC2_OFF + FC2_HALVES*2)

#define PREPACK_TOTAL (WIH_HALVES + WHH_HALVES + FC1_HALVES + FC2_HALVES + FC3_HALVES + BIASN)

__device__ __forceinline__ unsigned short f2bf(float f) {
  unsigned u = __float_as_uint(f);
  u += 0x7FFFu + ((u >> 16) & 1u);          // round-to-nearest-even
  return (unsigned short)(u >> 16);
}
__device__ __forceinline__ float sigf(float x) { return 1.0f / (1.0f + __expf(-x)); }
__device__ __forceinline__ float tanh_(float x) {
  float e = __expf(-2.0f * x);
  return (1.0f - e) / (1.0f + e);
}

// A fragment (16x32 bf16, M x K) from row-major LDS buffer.
// Lane l: m = l&15, hi = l>>4; halves j -> K = kbase+16*hi+j
__device__ __forceinline__ bf16x16 ldsA(const unsigned short* base, int stride, int kbase, int lane) {
  int m = lane & 15, hi = lane >> 4;
  const unsigned short* p = base + m * stride + kbase + hi * 16;
  u16x8 a0 = *(const u16x8*)p;          // ds_load_b128
  u16x8 a1 = *(const u16x8*)(p + 8);    // ds_load_b128
  u16x16 cat = __builtin_shufflevector(a0, a1, 0,1,2,3,4,5,6,7,8,9,10,11,12,13,14,15);
  return __builtin_bit_cast(bf16x16, cat);
}
// B fragment, prepacked per-lane contiguous (32B/lane) in global workspace (L2-resident)
__device__ __forceinline__ bf16x16 ldB(const unsigned short* tile, int lane) {
  return __builtin_bit_cast(bf16x16, *(const u16x16*)(tile + lane * 16));
}
__device__ __forceinline__ f32x8 wmma(bf16x16 a, bf16x16 b, f32x8 c) {
  return __builtin_amdgcn_wmma_f32_16x16x32_bf16(false, a, false, b, (short)0, c, false, false);
}

// ================= weight prepack: f32 row-major [N][K] -> bf16 WMMA-B fragments =================
__global__ void prepack_kernel(const float* __restrict__ w_ih, const float* __restrict__ w_hh,
                               const float* __restrict__ b_ih, const float* __restrict__ b_hh,
                               const float* __restrict__ fc1_w, const float* __restrict__ fc2_w,
                               const float* __restrict__ fc3_w, unsigned char* __restrict__ ws) {
  int i = blockIdx.x * 256 + threadIdx.x;
  if (i >= PREPACK_TOTAL) return;
  unsigned short* wihp = (unsigned short*)(ws + WIH_OFF);
  unsigned short* whhp = (unsigned short*)(ws + WHH_OFF);
  float*          bp   = (float*)(ws + BIAS_OFF);
  unsigned short* f1p  = (unsigned short*)(ws + FC1_OFF);
  unsigned short* f2p  = (unsigned short*)(ws + FC2_OFF);
  unsigned short* f3p  = (unsigned short*)(ws + FC3_OFF);

  if (i < WIH_HALVES) {   // w_ih: (L,2,256,128)
    int j = i & 15, lane = (i >> 4) & 31, nt = (i >> 9) & 15, kt = (i >> 13) & 3, ld = i >> 15;
    int n = nt * 16 + (lane & 15);
    int k = kt * 32 + (lane >> 4) * 16 + j;
    wihp[i] = f2bf(w_ih[(ld * 256 + n) * 128 + k]);
    return;
  }
  i -= WIH_HALVES;
  if (i < WHH_HALVES) {   // w_hh: (L,2,256,64)
    int j = i & 15, lane = (i >> 4) & 31, nt = (i >> 9) & 15, kt = (i >> 13) & 1, ld = i >> 14;
    int n = nt * 16 + (lane & 15);
    int k = kt * 32 + (lane >> 4) * 16 + j;
    whhp[i] = f2bf(w_hh[(ld * 256 + n) * 64 + k]);
    return;
  }
  i -= WHH_HALVES;
  if (i < FC1_HALVES) {   // fc1_w: (64,128)
    int j = i & 15, lane = (i >> 4) & 31, nt = (i >> 9) & 3, kt = i >> 11;
    int n = nt * 16 + (lane & 15);
    int k = kt * 32 + (lane >> 4) * 16 + j;
    f1p[i] = f2bf(fc1_w[n * 128 + k]);
    return;
  }
  i -= FC1_HALVES;
  if (i < FC2_HALVES) {   // fc2_w: (48,64)
    int kt = i / 1536, r = i - kt * 1536;
    int j = r & 15, lane = (r >> 4) & 31, nt = r >> 9;
    int n = nt * 16 + (lane & 15);
    int k = kt * 32 + (lane >> 4) * 16 + j;
    f2p[i] = f2bf(fc2_w[n * 64 + k]);
    return;
  }
  i -= FC2_HALVES;
  if (i < FC3_HALVES) {   // fc3_w: (32,48) with K zero-padded to 64
    int j = i & 15, lane = (i >> 4) & 31, nt = (i >> 9) & 1, kt = i >> 10;
    int n = nt * 16 + (lane & 15);
    int k = kt * 32 + (lane >> 4) * 16 + j;
    f3p[i] = f2bf(k < 48 ? fc3_w[n * 48 + k] : 0.0f);
    return;
  }
  i -= FC3_HALVES;
  bp[i] = b_ih[i] + b_hh[i];   // fused gate bias, f32
}

// ================= main: 16 batch rows per WG, 4 waves cooperate over gate N-tiles =================
__global__ __launch_bounds__(128) void lstm_mlp_kernel(
    const float* __restrict__ x, const unsigned char* __restrict__ ws,
    const float* __restrict__ fc1_b, const float* __restrict__ fc2_b,
    const float* __restrict__ fc3_b, const float* __restrict__ fc4_w,
    const float* __restrict__ fc4_b, float* __restrict__ out) {
  __shared__ __align__(32) unsigned short zA[STACKT * 16 * 128];   // 40 KB
  __shared__ __align__(32) unsigned short zB[STACKT * 16 * 128];   // 40 KB
  __shared__ __align__(32) unsigned short hbuf[16 * 64];           // recurrent h (bf16)
  __shared__ __align__(32) unsigned short fcbuf[16 * 64];          // fc2 out (padded)
  __shared__ __align__(32) float          fc3buf[16 * 32];

  const int tid = threadIdx.x;
  const int lane = tid & 31;
  const int wv = tid >> 5;          // wave 0..3 owns h columns [16*wv, 16*wv+16)
  const int n  = lane & 15;
  const int hi = lane >> 4;
  const int rowbase = blockIdx.x * 16;

  const unsigned short* wihp = (const unsigned short*)(ws + WIH_OFF);
  const unsigned short* whhp = (const unsigned short*)(ws + WHH_OFF);
  const float*          bp   = (const float*)(ws + BIAS_OFF);
  const unsigned short* f1p  = (const unsigned short*)(ws + FC1_OFF);
  const unsigned short* f2p  = (const unsigned short*)(ws + FC2_OFF);
  const unsigned short* f3p  = (const unsigned short*)(ws + FC3_OFF);

  // layer-0 input: x (B,40) -> z[t][row][0..3], zero-pad cols 4..127
  for (int idx = tid; idx < STACKT * 16 * 128; idx += 128) {
    int t = idx >> 11, rem = idx & 2047, r = rem >> 7, c = rem & 127;
    float v = (c < TIER) ? x[(rowbase + r) * (STACKT * TIER) + t * TIER + c] : 0.0f;
    zA[idx] = f2bf(v);
  }
  __syncthreads();

  unsigned short* zin = zA;
  unsigned short* zout = zB;

#pragma unroll 1
  for (int l = 0; l < NLAYER; ++l) {
#pragma unroll 1
    for (int d = 0; d < 2; ++d) {
      for (int idx = tid; idx < 16 * 64; idx += 128) hbuf[idx] = 0;
      f32x8 cfr;
#pragma unroll
      for (int r = 0; r < 8; ++r) cfr[r] = 0.0f;
      const unsigned short* wih_ld = wihp + (l * 2 + d) * (4 * 16 * 512);
      const unsigned short* whh_ld = whhp + (l * 2 + d) * (2 * 16 * 512);
      const float* bias_ld = bp + (l * 2 + d) * 256;
      __builtin_prefetch(wih_ld, 0, 0);   // global_prefetch of weight pack (L2-resident)
      __syncthreads();
#pragma unroll 1
      for (int s = 0; s < STACKT; ++s) {
        const int t = d ? (STACKT - 1 - s) : s;
        const unsigned short* zrow = zin + t * 16 * 128;
        // shared A fragments for all 4 gate tiles
        bf16x16 az0 = ldsA(zrow, 128, 0, lane);
        bf16x16 az1 = ldsA(zrow, 128, 32, lane);
        bf16x16 az2 = ldsA(zrow, 128, 64, lane);
        bf16x16 az3 = ldsA(zrow, 128, 96, lane);
        bf16x16 ah0 = ldsA(hbuf, 64, 0, lane);
        bf16x16 ah1 = ldsA(hbuf, 64, 32, lane);
        f32x8 g[4];
#pragma unroll
        for (int gi = 0; gi < 4; ++gi) {       // gi: 0=i 1=f 2=g 3=o ; N-tile = gi*4 + wv
          const int nt = gi * 4 + wv;
          float bv = bias_ld[nt * 16 + n];
          f32x8 acc;
#pragma unroll
          for (int r = 0; r < 8; ++r) acc[r] = bv;
          acc = wmma(az0, ldB(wih_ld + (0 * 16 + nt) * 512, lane), acc);
          acc = wmma(az1, ldB(wih_ld + (1 * 16 + nt) * 512, lane), acc);
          acc = wmma(az2, ldB(wih_ld + (2 * 16 + nt) * 512, lane), acc);
          acc = wmma(az3, ldB(wih_ld + (3 * 16 + nt) * 512, lane), acc);
          acc = wmma(ah0, ldB(whh_ld + (0 * 16 + nt) * 512, lane), acc);
          acc = wmma(ah1, ldB(whh_ld + (1 * 16 + nt) * 512, lane), acc);
          g[gi] = acc;
        }
        __syncthreads();   // all waves done reading hbuf before overwrite
#pragma unroll
        for (int r = 0; r < 8; ++r) {
          float iv = sigf(g[0][r]);
          float fv = sigf(g[1][r]);
          float gv = tanh_(g[2][r]);
          float ov = sigf(g[3][r]);
          float cv = fv * cfr[r] + iv * gv;
          cfr[r] = cv;
          unsigned short hb = f2bf(ov * tanh_(cv));
          int row = r + hi * 8;                 // C/D layout: vgpr r, lane half hi
          int col = wv * 16 + n;
          hbuf[row * 64 + col] = hb;
          zout[(t * 16 + row) * 128 + d * 64 + col] = hb;
        }
        __syncthreads();
      }
    }
    unsigned short* tmp = zin; zin = zout; zout = tmp;
  }

  // ---------------- MLP head ----------------
  const unsigned short* hlast = zin + (STACKT - 1) * 16 * 128;   // z[:, -1, :] (16 x 128)
  {  // fc1: 128 -> 64, wave wv does N-tile wv
    bf16x16 a0 = ldsA(hlast, 128, 0, lane);
    bf16x16 a1 = ldsA(hlast, 128, 32, lane);
    bf16x16 a2 = ldsA(hlast, 128, 64, lane);
    bf16x16 a3 = ldsA(hlast, 128, 96, lane);
    float bv = fc1_b[wv * 16 + n];
    f32x8 acc;
#pragma unroll
    for (int r = 0; r < 8; ++r) acc[r] = bv;
    acc = wmma(a0, ldB(f1p + (0 * 4 + wv) * 512, lane), acc);
    acc = wmma(a1, ldB(f1p + (1 * 4 + wv) * 512, lane), acc);
    acc = wmma(a2, ldB(f1p + (2 * 4 + wv) * 512, lane), acc);
    acc = wmma(a3, ldB(f1p + (3 * 4 + wv) * 512, lane), acc);
#pragma unroll
    for (int r = 0; r < 8; ++r)
      hbuf[(r + hi * 8) * 64 + wv * 16 + n] = f2bf(fmaxf(acc[r], 0.0f));
  }
  __syncthreads();
  if (wv < 3) {  // fc2: 64 -> 48
    bf16x16 a0 = ldsA(hbuf, 64, 0, lane);
    bf16x16 a1 = ldsA(hbuf, 64, 32, lane);
    float bv = fc2_b[wv * 16 + n];
    f32x8 acc;
#pragma unroll
    for (int r = 0; r < 8; ++r) acc[r] = bv;
    acc = wmma(a0, ldB(f2p + (0 * 3 + wv) * 512, lane), acc);
    acc = wmma(a1, ldB(f2p + (1 * 3 + wv) * 512, lane), acc);
#pragma unroll
    for (int r = 0; r < 8; ++r)
      fcbuf[(r + hi * 8) * 64 + wv * 16 + n] = f2bf(fmaxf(acc[r], 0.0f));
  } else {       // wave 3 zero-pads cols 48..63 (K padding for fc3)
    for (int idx = lane; idx < 256; idx += 32) {
      int r = idx >> 4, c = idx & 15;
      fcbuf[r * 64 + 48 + c] = 0;
    }
  }
  __syncthreads();
  if (wv < 2) {  // fc3: 48(pad 64) -> 32
    bf16x16 a0 = ldsA(fcbuf, 64, 0, lane);
    bf16x16 a1 = ldsA(fcbuf, 64, 32, lane);
    float bv = fc3_b[wv * 16 + n];
    f32x8 acc;
#pragma unroll
    for (int r = 0; r < 8; ++r) acc[r] = bv;
    acc = wmma(a0, ldB(f3p + (0 * 2 + wv) * 512, lane), acc);
    acc = wmma(a1, ldB(f3p + (1 * 2 + wv) * 512, lane), acc);
#pragma unroll
    for (int r = 0; r < 8; ++r)
      fc3buf[(r + hi * 8) * 32 + wv * 16 + n] = fmaxf(acc[r], 0.0f);
  }
  __syncthreads();
  if (tid < 16) {  // fc4: 32 -> 1, relu
    float s = fc4_b[0];
#pragma unroll
    for (int k = 0; k < 32; ++k) s += fc3buf[tid * 32 + k] * fc4_w[k];
    out[rowbase + tid] = fmaxf(s, 0.0f);
  }
}

extern "C" void kernel_launch(void* const* d_in, const int* in_sizes, int n_in,
                              void* d_out, int out_size, void* d_ws, size_t ws_size,
                              hipStream_t stream) {
  const float* x     = (const float*)d_in[0];
  // d_in[1] = y (unused by reference)
  const float* w_ih  = (const float*)d_in[2];
  const float* w_hh  = (const float*)d_in[3];
  const float* b_ih  = (const float*)d_in[4];
  const float* b_hh  = (const float*)d_in[5];
  const float* fc1_w = (const float*)d_in[6];
  const float* fc1_b = (const float*)d_in[7];
  const float* fc2_w = (const float*)d_in[8];
  const float* fc2_b = (const float*)d_in[9];
  const float* fc3_w = (const float*)d_in[10];
  const float* fc3_b = (const float*)d_in[11];
  const float* fc4_w = (const float*)d_in[12];
  const float* fc4_b = (const float*)d_in[13];
  unsigned char* ws  = (unsigned char*)d_ws;
  float* out         = (float*)d_out;

  const int B = in_sizes[0] / (STACKT * TIER);   // 32768

  int pre_blocks = (PREPACK_TOTAL + 255) / 256;
  prepack_kernel<<<pre_blocks, 256, 0, stream>>>(w_ih, w_hh, b_ih, b_hh,
                                                 fc1_w, fc2_w, fc3_w, ws);
  lstm_mlp_kernel<<<B / 16, 128, 0, stream>>>(x, ws, fc1_b, fc2_b, fc3_b,
                                              fc4_w, fc4_b, out);
}
